// VectorQuantizer_17755394801832
// MI455X (gfx1250) — matrix operand order repaired and checked by reading, not compiled, not run on hardware
//
#include <hip/hip_runtime.h>
#include <hip/hip_bf16.h>

// ---------------------------------------------------------------------------
// Problem sizes (fixed by the reference)
// ---------------------------------------------------------------------------
#define VQ_N 65536
#define VQ_K 1024
#define VQ_D 256

typedef __attribute__((ext_vector_type(4)))  float        f32x4;
typedef __attribute__((ext_vector_type(4)))  unsigned int u32x4;
typedef __attribute__((ext_vector_type(8)))  __bf16       v8bf;
typedef __attribute__((ext_vector_type(16))) __bf16       v16bf;
typedef __attribute__((ext_vector_type(8)))  float        v8f;

// ---------------------------------------------------------------------------
// CDNA5 async copy: global -> LDS without touching VGPR data paths.
// Tracked with ASYNCcnt; each lane's LDS address comes from lds_off.
// ---------------------------------------------------------------------------
__device__ __forceinline__ void vq_async_ld2(unsigned lds_off, const __bf16* g)
{
    // two 16-byte chunks per thread (32 B total)
    asm volatile(
        "global_load_async_to_lds_b128 %0, %2, off\n\t"
        "global_load_async_to_lds_b128 %1, %3, off"
        :
        : "v"(lds_off), "v"(lds_off + 16u), "v"(g), "v"(g + 8)
        : "memory");
}

__device__ __forceinline__ void vq_wait_async0()
{
    asm volatile("s_wait_asynccnt 0x0" ::: "memory");
}

// ---------------------------------------------------------------------------
// Kernel 1: codebook prep.
//   - bf16 copy of weight (WMMA A-operand source, stays hot in L2: 512 KB)
//   - half_wnorm[k] = 0.5 * ||w_k||^2  (f32, folded into argmax score)
//   - zero the histogram counters
// One block (256 threads) per code row.
// ---------------------------------------------------------------------------
__global__ __launch_bounds__(256) void vq_prep_kernel(
    const float* __restrict__ weight,
    __bf16* __restrict__ wbf,
    float* __restrict__ half_wnorm,
    unsigned int* __restrict__ counts)
{
    const int k = blockIdx.x;
    const int d = threadIdx.x;
    const float w = weight[k * VQ_D + d];
    wbf[k * VQ_D + d] = (__bf16)w;

    float p = w * w;
#pragma unroll
    for (int off = 16; off > 0; off >>= 1) p += __shfl_down(p, off, 32);

    __shared__ float sw[8];
    if ((threadIdx.x & 31) == 0) sw[threadIdx.x >> 5] = p;
    __syncthreads();
    if (threadIdx.x == 0) {
        float s = 0.0f;
#pragma unroll
        for (int i = 0; i < 8; ++i) s += sw[i];
        half_wnorm[k] = 0.5f * s;
    }
    if (blockIdx.x < VQ_K / 256) counts[blockIdx.x * 256 + threadIdx.x] = 0u;
}

// ---------------------------------------------------------------------------
// Load one 16x32 bf16 chunk of the e-tile as a WMMA B operand (this code's contiguous k order, same as A).
// B as used here (32x16, KxN): lane n (0-15) holds col N=n, K = dc*32 + 0..15;
// lane n+16 holds col N=n, K = dc*32 + 16..31 (packed 2 bf16 / VGPR).
// ---------------------------------------------------------------------------
__device__ __forceinline__ v16bf vq_load_b_chunk(const float* erow, int dc, int hs)
{
    const f32x4* p = (const f32x4*)(erow + dc * 32 + hs * 16);
    f32x4 x[4];
    x[0] = p[0]; x[1] = p[1]; x[2] = p[2]; x[3] = p[3];
    v16bf b;
#pragma unroll
    for (int j = 0; j < 16; ++j) b[j] = (__bf16)x[j >> 2][j & 3];
    return b;
}

// ---------------------------------------------------------------------------
// Kernel 2: WMMA distance + argmin.
// Grid: N/256 blocks of 256 threads (8 waves). Each wave owns 32 e-rows
// (two 16-row B register sets -> two independent WMMA accumulator chains,
// sharing every A-tile fetch), and sweeps all 64 codebook tiles
// (16 codes x 256 dims). Tiles are staged in a double-buffered LDS buffer
// via GLOBAL_LOAD_ASYNC_TO_LDS_B128 (ASYNCcnt) shared by the workgroup.
//   score = dot(e_n, w_k) - 0.5*||w_k||^2 ; argmax(score) == argmin(dist)
// ---------------------------------------------------------------------------
__global__ __launch_bounds__(256) void vq_argmin_kernel(
    const float* __restrict__ e,
    const __bf16* __restrict__ wbf,
    const float* __restrict__ half_wnorm,
    float* __restrict__ out_idx_f,
    int* __restrict__ idx_i32)
{
    __shared__ alignas(16) __bf16 s_codes[2][16 * VQ_D]; // 2 x 8 KB
    __shared__ float s_hw[VQ_K];                          // 4 KB

    const int tid  = threadIdx.x;
    const int lane = tid & 31;
    const int wv   = tid >> 5;
    const int nloc = lane & 15;   // e-row / code-in-tile selector
    const int hs   = lane >> 4;   // half-select (K-halves of the A/B layouts)
    const int rowBase = blockIdx.x * 256 + wv * 32;
    const int row0 = rowBase + nloc;
    const int row1 = rowBase + 16 + nloc;

    // stage half_wnorm once
    for (int i = tid; i < VQ_K; i += 256) s_hw[i] = half_wnorm[i];

    // ---- two 16x256 e-tiles in registers as bf16 (B operands)
    v16bf B0[8], B1[8];
    {
        const float* erow0 = e + (size_t)row0 * VQ_D;
        const float* erow1 = e + (size_t)row1 * VQ_D;
#pragma unroll
        for (int dc = 0; dc < 8; ++dc) B0[dc] = vq_load_b_chunk(erow0, dc, hs);
#pragma unroll
        for (int dc = 0; dc < 8; ++dc) B1[dc] = vq_load_b_chunk(erow1, dc, hs);
    }

    // LDS byte offsets of the two code buffers (flat-LDS keeps offset in [31:0])
    const unsigned ldsBuf0 = (unsigned)(size_t)(&s_codes[0][0]) + (unsigned)tid * 32u;
    const unsigned ldsBuf1 = (unsigned)(size_t)(&s_codes[1][0]) + (unsigned)tid * 32u;

    // ---- async preload of codebook tile 0 (8 KB, 32 B / thread)
    vq_async_ld2(ldsBuf0, wbf + tid * 16);
    vq_wait_async0();
    __syncthreads();

    float best0S = -3.4e38f, best1S = -3.4e38f;
    int   best0C = 0,        best1C = 0;

    for (int kt = 0; kt < VQ_K / 16; ++kt) {
        const int cur = kt & 1;

        // async-prefetch the next 16-code tile into the other buffer
        if (kt + 1 < VQ_K / 16) {
            vq_async_ld2(cur ? ldsBuf0 : ldsBuf1,
                         wbf + (size_t)(kt + 1) * 16 * VQ_D + tid * 16);
        }

        // ---- 16 WMMAs: 8 per accumulator chain, A-tile shared by both
        // A as used here (16x32, contiguous k): lane m (0-15): code M=m, dims {0..15};
        // lane m+16: code M=m, dims {16..31} -> two ds_load_b128 each.
        v8f c0 = {}, c1 = {};
        const __bf16* abase = &s_codes[cur][nloc * VQ_D + hs * 16];
#pragma unroll
        for (int dc = 0; dc < 8; ++dc) {
            const v8bf* ap = (const v8bf*)(abase + dc * 32);
            v8bf lo = ap[0];
            v8bf hi = ap[1];
            v16bf a = __builtin_shufflevector(lo, hi, 0, 1, 2, 3, 4, 5, 6, 7,
                                              8, 9, 10, 11, 12, 13, 14, 15);
            c0 = __builtin_amdgcn_wmma_f32_16x16x32_bf16(
                     false, a, false, B0[dc], (short)0, c0, false, false);
            c1 = __builtin_amdgcn_wmma_f32_16x16x32_bf16(
                     false, a, false, B1[dc], (short)0, c1, false, false);
        }

        // C layout: lanes 0-15 hold M=0..7 (VGPR r) for N=lane;
        // lanes 16-31 hold M=8..15 for N=lane-16.
        const int codeBase = kt * 16 + hs * 8;
        const float* hw = &s_hw[codeBase];
#pragma unroll
        for (int r = 0; r < 8; ++r) {
            const float h = hw[r];
            const float s0 = c0[r] - h;
            const float s1 = c1[r] - h;
            if (s0 > best0S) { best0S = s0; best0C = codeBase + r; }
            if (s1 > best1S) { best1S = s1; best1C = codeBase + r; }
        }

        // our async loads (into the other buffer) must land before next iter
        vq_wait_async0();
        __syncthreads();
    }

    // merge lane n with lane n+16 (they hold complementary 8-code halves);
    // tie-break to the lower index to match argmin's first-match rule.
    {
        const float oS = __shfl_xor(best0S, 16, 32);
        const int   oC = __shfl_xor(best0C, 16, 32);
        if (oS > best0S || (oS == best0S && oC < best0C)) { best0S = oS; best0C = oC; }
    }
    {
        const float oS = __shfl_xor(best1S, 16, 32);
        const int   oC = __shfl_xor(best1C, 16, 32);
        if (oS > best1S || (oS == best1S && oC < best1C)) { best1S = oS; best1C = oC; }
    }

    if (hs == 0) {
        out_idx_f[row0] = (float)best0C;
        idx_i32[row0]   = best0C;
        out_idx_f[row1] = (float)best1C;
        idx_i32[row1]   = best1C;
    }
}

// ---------------------------------------------------------------------------
// Kernel 3: gather quantized = weight[idx], write output, per-block loss
// partial sums (deterministic: fixed-order tree reduction, one partial per
// block, no float atomics). One thread = one float4 of the N x D grid.
// ---------------------------------------------------------------------------
__global__ __launch_bounds__(256) void vq_gather_kernel(
    const float* __restrict__ e,
    const float* __restrict__ weight,
    const int* __restrict__ idx_i32,
    float* __restrict__ qout,
    float* __restrict__ partials)
{
    const int g    = blockIdx.x * 256 + threadIdx.x; // N*D/4 groups
    const int rowi = g >> 6;                         // D/4 = 64 groups/row
    const int c4   = g & 63;
    const int k    = idx_i32[rowi];

    const f32x4 q  = ((const f32x4*)weight)[k * (VQ_D / 4) + c4];
    const f32x4 ev = ((const f32x4*)e)[(size_t)rowi * (VQ_D / 4) + c4];
    ((f32x4*)qout)[(size_t)rowi * (VQ_D / 4) + c4] = q; // straight-through fwd

    const f32x4 d = q - ev;
    float s = d[0] * d[0] + d[1] * d[1] + d[2] * d[2] + d[3] * d[3];
#pragma unroll
    for (int off = 16; off > 0; off >>= 1) s += __shfl_down(s, off, 32);

    __shared__ float sw[8];
    if ((threadIdx.x & 31) == 0) sw[threadIdx.x >> 5] = s;
    __syncthreads();
    if (threadIdx.x == 0) {
        float t = 0.0f;
#pragma unroll
        for (int i = 0; i < 8; ++i) t += sw[i];
        partials[blockIdx.x] = t;
    }
}

// ---------------------------------------------------------------------------
// Kernel 4: histogram (integer atomics -> deterministic)
// ---------------------------------------------------------------------------
__global__ __launch_bounds__(256) void vq_hist_kernel(
    const int* __restrict__ idx_i32, unsigned int* __restrict__ counts)
{
    const int i = blockIdx.x * 256 + threadIdx.x;
    atomicAdd(&counts[idx_i32[i]], 1u);
}

// ---------------------------------------------------------------------------
// Kernel 5: finalize scalars: commitment_loss, codebook_usage
// ---------------------------------------------------------------------------
__global__ __launch_bounds__(256) void vq_finalize_kernel(
    const unsigned int* __restrict__ counts,
    const float* __restrict__ partials,
    int n_partials,
    float* __restrict__ out2) // out2[0]=loss, out2[1]=usage
{
    const int tid = threadIdx.x;
    unsigned int mx = 0u, sm = 0u;
    for (int i = tid; i < VQ_K; i += 256) {
        const unsigned int cnt = counts[i];
        mx = (cnt > mx) ? cnt : mx;
        sm += cnt;
    }
    float ls = 0.0f;
    for (int i = tid; i < n_partials; i += 256) ls += partials[i];

    __shared__ unsigned int smx[256];
    __shared__ unsigned int ssm[256];
    __shared__ float        sls[256];
    smx[tid] = mx; ssm[tid] = sm; sls[tid] = ls;
    __syncthreads();
    for (int s = 128; s > 0; s >>= 1) {
        if (tid < s) {
            smx[tid] = (smx[tid + s] > smx[tid]) ? smx[tid + s] : smx[tid];
            ssm[tid] += ssm[tid + s];
            sls[tid] += sls[tid + s];
        }
        __syncthreads();
    }
    if (tid == 0) {
        out2[0] = sls[0] / (float)((size_t)VQ_N * VQ_D); // commitment_loss
        out2[1] = (float)smx[0] / (float)ssm[0];         // codebook_usage
    }
}

// ---------------------------------------------------------------------------
// Host-side launch. d_in[0] = e [N,D] f32, d_in[1] = weight [K,D] f32.
// d_out = [quantized NxD | indices N (as f32) | loss | usage].
// ---------------------------------------------------------------------------
extern "C" void kernel_launch(void* const* d_in, const int* in_sizes, int n_in,
                              void* d_out, int out_size, void* d_ws, size_t ws_size,
                              hipStream_t stream)
{
    (void)in_sizes; (void)n_in; (void)out_size; (void)ws_size;

    const float* e      = (const float*)d_in[0];
    const float* weight = (const float*)d_in[1];
    float* outf = (float*)d_out;

    // workspace carve-up (all 16B aligned)
    char* ws = (char*)d_ws;
    __bf16*       wbf      = (__bf16*)ws;                         // 512 KB
    float*        hwn      = (float*)(ws + 524288);               //   4 KB
    int*          idx      = (int*)(ws + 528384);                 // 256 KB
    unsigned int* counts   = (unsigned int*)(ws + 790528);        //   4 KB
    float*        partials = (float*)(ws + 794624);               //  64 KB

    const int n_gather_blocks = (VQ_N * VQ_D / 4) / 256; // 16384

    vq_prep_kernel<<<VQ_K, 256, 0, stream>>>(weight, wbf, hwn, counts);
    vq_argmin_kernel<<<VQ_N / 256, 256, 0, stream>>>(
        e, wbf, hwn, outf + (size_t)VQ_N * VQ_D, idx);
    vq_gather_kernel<<<n_gather_blocks, 256, 0, stream>>>(
        e, weight, idx, outf, partials);
    vq_hist_kernel<<<VQ_N / 256, 256, 0, stream>>>(idx, counts);
    vq_finalize_kernel<<<1, 256, 0, stream>>>(
        counts, partials, n_gather_blocks,
        outf + (size_t)VQ_N * VQ_D + VQ_N);
}
